// GRU_55825984914083
// MI455X (gfx1250) — compile-verified, matrix operand order checked by reading, not run on hardware
//
#include <hip/hip_runtime.h>
#include <hip/hip_bf16.h>

// ---------------------------------------------------------------------------
// GRU on gfx1250: bf16 WMMA (16x16x32) GEMM tiles, f32 accumulation.
// dims: B=256, T=512, D=128, Din=257 (padded->288), H=1024, O=128
// ---------------------------------------------------------------------------

typedef __attribute__((ext_vector_type(16))) __bf16 v16bf;
typedef __attribute__((ext_vector_type(8)))  __bf16 v8bf;
typedef __attribute__((ext_vector_type(8)))  float  v8f;

#define GRU_B    256
#define GRU_T    512
#define GRU_D    128
#define GRU_DIN  257
#define GRU_DINP 288   // 257 padded up to a multiple of 32
#define GRU_H    1024
#define GRU_O    128

static __device__ __forceinline__ unsigned short f32_to_bf16_rne(float f) {
    unsigned int u = __float_as_uint(f);
    u += 0x7FFFu + ((u >> 16) & 1u);   // round-to-nearest-even
    return (unsigned short)(u >> 16);
}
static __device__ __forceinline__ float bf16_to_f32(unsigned short s) {
    return __uint_as_float(((unsigned int)s) << 16);
}

// Load one 16x32 bf16 operand tile (A layout; B uses the same pattern when the
// weight matrix is stored [N][K] row-major). `base` points at (row0, k0),
// `ld` is the row stride in elements. Requires 16-byte alignment (guaranteed:
// all strides are multiples of 8 elements and k steps by 32).
static __device__ __forceinline__ v16bf load_tile_16x32(
        const unsigned short* __restrict__ base, int ld, int lane) {
    const int row = lane & 15;
    const int kg  = lane >> 4;                 // 0 or 1
    const unsigned short* p = base + row * ld + kg * 8;
    v8bf lo = *(const v8bf*)(p);               // K = kg*8 .. kg*8+7
    v8bf hi = *(const v8bf*)(p + 16);          // K = kg*8+16 .. kg*8+23
    return __builtin_shufflevector(lo, hi, 0,1,2,3,4,5,6,7,
                                           8,9,10,11,12,13,14,15);
}

static __device__ __forceinline__ v8f wmma_bf16(v16bf a, v16bf b, v8f c) {
    return __builtin_amdgcn_wmma_f32_16x16x32_bf16(
        /*neg_a=*/false, a, /*neg_b=*/false, b,
        /*c_mod=*/(short)0, c, /*reuse_a=*/false, /*reuse_b=*/false);
}

static __device__ __forceinline__ float fast_sigmoid(float x) {
    return 1.0f / (1.0f + __expf(-x));
}

// ---------------------------------------------------------------------------
// Pack kernels
// ---------------------------------------------------------------------------

// f32 [rows][scols] -> bf16 [rows][dcols], zero padded.
__global__ void pack_pad_bf16(const float* __restrict__ src,
                              unsigned short* __restrict__ dst,
                              int rows, int scols, int dcols) {
    int n = rows * dcols;
    for (int i = blockIdx.x * blockDim.x + threadIdx.x; i < n;
         i += gridDim.x * blockDim.x) {
        int r = i / dcols, k = i - r * dcols;
        dst[i] = (k < scols) ? f32_to_bf16_rne(src[r * scols + k])
                             : (unsigned short)0;
    }
}

// Build time-major packed features: xt[t][b][k], k in [0,288)
//   k <  128 : x[b][t][k]
//   k <  256 : mask[b][t][k-128]
//   k == 256 : ti[b][t]
//   else     : 0
__global__ void pack_xt_bf16(const float* __restrict__ x,
                             const float* __restrict__ mask,
                             const float* __restrict__ ti,
                             unsigned short* __restrict__ xt) {
    const int n = GRU_T * GRU_B * GRU_DINP;
    for (int i = blockIdx.x * blockDim.x + threadIdx.x; i < n;
         i += gridDim.x * blockDim.x) {
        int k  = i % GRU_DINP;
        int tb = i / GRU_DINP;
        int b  = tb % GRU_B;
        int t  = tb / GRU_B;
        size_t bt = (size_t)b * GRU_T + t;
        float v;
        if (k < GRU_D)            v = x[bt * GRU_D + k];
        else if (k < 2 * GRU_D)   v = mask[bt * GRU_D + (k - GRU_D)];
        else if (k == 2 * GRU_D)  v = ti[bt];
        else                      v = 0.0f;
        xt[i] = f32_to_bf16_rne(v);
    }
}

__global__ void zero_bf16(unsigned short* __restrict__ p, int n) {
    for (int i = blockIdx.x * blockDim.x + threadIdx.x; i < n;
         i += gridDim.x * blockDim.x)
        p[i] = 0;
}

// ---------------------------------------------------------------------------
// One GRU time step. Grid: (H/16/4, B/16) = (16,16), block: 128 (4 waves).
// Wave w computes the 16x16 tile (b0..b0+15) x (j0..j0+15) of h_new.
//   gates: r = sigmoid(i_r + h_r + biases)
//          z = sigmoid(i_z + h_z + biases)
//          n = tanh(i_n + b_ihn + r * (h_n + b_hhn))
//   h'   = (1-z)*n + z*h
// ---------------------------------------------------------------------------
__global__ __launch_bounds__(128)
void gru_step(const unsigned short* __restrict__ hprev,   // [B][H] bf16
              unsigned short* __restrict__ hnext,         // [B][H] bf16
              const unsigned short* __restrict__ xt,      // [B][288] bf16 (this t)
              const unsigned short* __restrict__ Wih,     // [3H][288] bf16
              const unsigned short* __restrict__ Whh,     // [3H][H] bf16
              const float* __restrict__ b_ih,             // [3H]
              const float* __restrict__ b_hh) {           // [3H]
    const int lane = threadIdx.x & 31;
    const int wid  = threadIdx.x >> 5;
    const int j0 = (blockIdx.x * 4 + wid) * 16;   // hidden-unit tile
    const int b0 = blockIdx.y * 16;               // batch tile

    v8f accR  = {};   // i_r + h_r
    v8f accZ  = {};   // i_z + h_z
    v8f accIN = {};   // i_n
    v8f accHN = {};   // h_n

    // ---- hidden-side projections: K = H = 1024 ----
    {
        const unsigned short* A  = hprev + (size_t)b0 * GRU_H;
        const unsigned short* Br = Whh + (size_t)(0 * GRU_H + j0) * GRU_H;
        const unsigned short* Bz = Whh + (size_t)(1 * GRU_H + j0) * GRU_H;
        const unsigned short* Bn = Whh + (size_t)(2 * GRU_H + j0) * GRU_H;
        #pragma unroll 4
        for (int k = 0; k < GRU_H; k += 32) {
            v16bf a  = load_tile_16x32(A + k, GRU_H, lane);
            v16bf br = load_tile_16x32(Br + k, GRU_H, lane);
            accR  = wmma_bf16(a, br, accR);
            v16bf bz = load_tile_16x32(Bz + k, GRU_H, lane);
            accZ  = wmma_bf16(a, bz, accZ);
            v16bf bn = load_tile_16x32(Bn + k, GRU_H, lane);
            accHN = wmma_bf16(a, bn, accHN);
        }
    }

    // ---- input-side projections: K = DinP = 288 ----
    {
        const unsigned short* A  = xt + (size_t)b0 * GRU_DINP;
        const unsigned short* Br = Wih + (size_t)(0 * GRU_H + j0) * GRU_DINP;
        const unsigned short* Bz = Wih + (size_t)(1 * GRU_H + j0) * GRU_DINP;
        const unsigned short* Bn = Wih + (size_t)(2 * GRU_H + j0) * GRU_DINP;
        #pragma unroll
        for (int k = 0; k < GRU_DINP; k += 32) {
            v16bf a  = load_tile_16x32(A + k, GRU_DINP, lane);
            v16bf br = load_tile_16x32(Br + k, GRU_DINP, lane);
            accR  = wmma_bf16(a, br, accR);
            v16bf bz = load_tile_16x32(Bz + k, GRU_DINP, lane);
            accZ  = wmma_bf16(a, bz, accZ);
            v16bf bn = load_tile_16x32(Bn + k, GRU_DINP, lane);
            accIN = wmma_bf16(a, bn, accIN);
        }
    }

    // ---- epilogue: activations + convex update ----
    const int col   = lane & 15;
    const int rbase = (lane >> 4) * 8;            // C/D layout row base
    const int j     = j0 + col;
    const float biasR  = b_ih[j]             + b_hh[j];
    const float biasZ  = b_ih[GRU_H + j]     + b_hh[GRU_H + j];
    const float biasIN = b_ih[2 * GRU_H + j];
    const float biasHN = b_hh[2 * GRU_H + j];

    #pragma unroll
    for (int v = 0; v < 8; ++v) {
        const int b = b0 + rbase + v;
        float r  = fast_sigmoid(accR[v] + biasR);
        float z  = fast_sigmoid(accZ[v] + biasZ);
        float n  = tanhf(accIN[v] + biasIN + r * (accHN[v] + biasHN));
        float hp = bf16_to_f32(hprev[(size_t)b * GRU_H + j]);
        float hn = (1.0f - z) * n + z * hp;
        hnext[(size_t)b * GRU_H + j] = f32_to_bf16_rne(hn);
    }
}

// ---------------------------------------------------------------------------
// out = h_last @ W_out.T + b_out   -> f32 [B][O]
// Grid: (O/16, B/16) = (8,16), block: 32 (one wave per 16x16 tile).
// ---------------------------------------------------------------------------
__global__ __launch_bounds__(32)
void out_gemm(const unsigned short* __restrict__ h,     // [B][H] bf16
              const unsigned short* __restrict__ Wout,  // [O][H] bf16
              const float* __restrict__ b_out,          // [O]
              float* __restrict__ out) {                // [B][O] f32
    const int lane = threadIdx.x & 31;
    const int o0 = blockIdx.x * 16;
    const int b0 = blockIdx.y * 16;

    v8f acc = {};
    const unsigned short* A = h    + (size_t)b0 * GRU_H;
    const unsigned short* Bm = Wout + (size_t)o0 * GRU_H;
    #pragma unroll 4
    for (int k = 0; k < GRU_H; k += 32) {
        v16bf a = load_tile_16x32(A + k, GRU_H, lane);
        v16bf b = load_tile_16x32(Bm + k, GRU_H, lane);
        acc = wmma_bf16(a, b, acc);
    }

    const int col   = lane & 15;
    const int rbase = (lane >> 4) * 8;
    const float bias = b_out[o0 + col];
    #pragma unroll
    for (int v = 0; v < 8; ++v) {
        const int b = b0 + rbase + v;
        out[(size_t)b * GRU_O + o0 + col] = acc[v] + bias;
    }
}

// ---------------------------------------------------------------------------
// Host launcher
// ---------------------------------------------------------------------------
extern "C" void kernel_launch(void* const* d_in, const int* in_sizes, int n_in,
                              void* d_out, int out_size, void* d_ws, size_t ws_size,
                              hipStream_t stream) {
    (void)in_sizes; (void)n_in; (void)out_size; (void)ws_size;

    const float* x     = (const float*)d_in[0];  // [256,512,128]
    const float* mask  = (const float*)d_in[1];  // [256,512,128]
    const float* ti    = (const float*)d_in[2];  // [256,512,1]
    const float* W_ih  = (const float*)d_in[3];  // [3072,257]
    const float* W_hh  = (const float*)d_in[4];  // [3072,1024]
    const float* b_ih  = (const float*)d_in[5];  // [3072]
    const float* b_hh  = (const float*)d_in[6];  // [3072]
    const float* W_out = (const float*)d_in[7];  // [128,1024]
    const float* b_out = (const float*)d_in[8];  // [128]
    float* out = (float*)d_out;                  // [256,128]

    // Workspace layout (bf16 elements), all 16B aligned.
    char* ws = (char*)d_ws;
    size_t off = 0;
    unsigned short* wih  = (unsigned short*)(ws + off); off += (size_t)3 * GRU_H * GRU_DINP * 2; // 1.77 MB
    unsigned short* whh  = (unsigned short*)(ws + off); off += (size_t)3 * GRU_H * GRU_H    * 2; // 6.29 MB
    unsigned short* wout = (unsigned short*)(ws + off); off += (size_t)GRU_O * GRU_H        * 2; // 0.26 MB
    unsigned short* xt   = (unsigned short*)(ws + off); off += (size_t)GRU_T * GRU_B * GRU_DINP * 2; // 75.5 MB
    unsigned short* h0   = (unsigned short*)(ws + off); off += (size_t)GRU_B * GRU_H * 2;    // 0.5 MB
    unsigned short* h1   = (unsigned short*)(ws + off); off += (size_t)GRU_B * GRU_H * 2;    // 0.5 MB

    const int thr = 256;
    // Pack weights.
    {
        int n = 3 * GRU_H * GRU_DINP;
        pack_pad_bf16<<<(n + thr - 1) / thr, thr, 0, stream>>>(W_ih, wih, 3 * GRU_H, GRU_DIN, GRU_DINP);
    }
    {
        int n = 3 * GRU_H * GRU_H;
        pack_pad_bf16<<<(n + thr - 1) / thr, thr, 0, stream>>>(W_hh, whh, 3 * GRU_H, GRU_H, GRU_H);
    }
    {
        int n = GRU_O * GRU_H;
        pack_pad_bf16<<<(n + thr - 1) / thr, thr, 0, stream>>>(W_out, wout, GRU_O, GRU_H, GRU_H);
    }
    // Pack features (time-major, padded, bf16).
    {
        int n = GRU_T * GRU_B * GRU_DINP;
        pack_xt_bf16<<<(n + thr - 1) / thr, thr, 0, stream>>>(x, mask, ti, xt);
    }
    // h0 = 0.
    {
        int n = GRU_B * GRU_H;
        zero_bf16<<<(n + thr - 1) / thr, thr, 0, stream>>>(h0, n);
    }

    // Sequential scan: one kernel per step (implicit global sync between steps).
    unsigned short* hb[2] = { h0, h1 };
    dim3 sgrid(GRU_H / 16 / 4, GRU_B / 16);   // (16,16) blocks, 4 waves each
    for (int t = 0; t < GRU_T; ++t) {
        gru_step<<<sgrid, 128, 0, stream>>>(
            hb[t & 1], hb[(t + 1) & 1],
            xt + (size_t)t * GRU_B * GRU_DINP,
            wih, whh, b_ih, b_hh);
    }
    // After 512 steps, h_last lives in hb[0].
    out_gemm<<<dim3(GRU_O / 16, GRU_B / 16), 32, 0, stream>>>(hb[0], wout, b_out, out);
}
